// DeepAR_62689342652886
// MI455X (gfx1250) — compile-verified, matrix operand order checked by reading, not run on hardware
//
#include <hip/hip_runtime.h>
#include <hip/hip_bf16.h>
#include <cstdint>

typedef __attribute__((ext_vector_type(16))) _Float16 v16h;
typedef __attribute__((ext_vector_type(8)))  _Float16 v8h;
typedef __attribute__((ext_vector_type(8)))  float    v8f;

#define HIST   336
#define CNTX   168
#define PRED   24
#define HID    128
#define NFEAT  5
#define EMB    50
#define NLAGS  168
#define IN_DIM 218
#define NSAMP  8192

// Fused-GEMM K layout: [0,168) lags | [168,218) ym | [218,224) zero pad | [224,352) h
#define KTOT 352
#define KH   224
#define XSTR 360   // LDS A-row stride in halves (720 B, 16B aligned)
#define MBLK 32    // sample rows per workgroup
#define BUFS 192   // NLAGS + PRED lag ring length

// workspace byte offsets (total ~336 KB)
#define WS_SC     0
#define WS_INV    4
#define WS_HLAST  16      // 128 f32
#define WS_YM     1024    // 24*50 f32
#define WS_BB     8192    // 4*128 f32: b_r(ih+hh), b_z(ih+hh), b_ih_n, b_hh_n
#define WS_WN     16384   // 128*128 f16: W_hh n-gate, [n][k]
#define WS_WC     65536   // 384*352 f16: combined weights, [n][k]

__device__ __forceinline__ v16h load_frag(const _Float16* p) {
    // 16-bit WMMA fragment for this lane: halves {k..k+7, k+16..k+23}
    v8h lo = *(const v8h*)(p);
    v8h hh = *(const v8h*)(p + 16);
    return __builtin_shufflevector(lo, hh, 0,1,2,3,4,5,6,7,8,9,10,11,12,13,14,15);
}

__device__ __forceinline__ v8f wmma16(v16h a, v16h b, v8f c) {
    return __builtin_amdgcn_wmma_f32_16x16x32_f16(false, a, false, b, (short)0, c, false, false);
}

__device__ __forceinline__ uint64_t splitmix64(uint64_t z) {
    z += 0x9E3779B97F4A7C15ull;
    z = (z ^ (z >> 30)) * 0xBF58476D1CE4E5B9ull;
    z = (z ^ (z >> 27)) * 0x94D049BB133111EBull;
    return z ^ (z >> 31);
}

__device__ __forceinline__ float softplusf(float v) {
    return v > 20.f ? v : log1pf(__expf(v));
}

// ---------------------------------------------------------------- prep ----
__global__ __launch_bounds__(256) void deepar_prep(
    const float* __restrict__ y_mark, const float* __restrict__ W_emb,
    const float* __restrict__ b_emb,  const float* __restrict__ W_ih,
    const float* __restrict__ W_hh,   const float* __restrict__ b_ih,
    const float* __restrict__ b_hh,   char* __restrict__ ws)
{
    _Float16* Wc = (_Float16*)(ws + WS_WC);
    _Float16* Wn = (_Float16*)(ws + WS_WN);
    float*    bb = (float*)(ws + WS_BB);
    float*    ym = (float*)(ws + WS_YM);
    const int gid = blockIdx.x * blockDim.x + threadIdx.x;
    const int stride = gridDim.x * blockDim.x;

    for (int i = gid; i < 3 * HID * KTOT; i += stride) {
        int n = i / KTOT, k = i - n * KTOT;
        float v;
        if (k < IN_DIM)       v = W_ih[n * IN_DIM + k];
        else if (k < KH)      v = 0.f;
        else                  v = (n < 2 * HID) ? W_hh[n * HID + (k - KH)] : 0.f;
        Wc[i] = (_Float16)v;
    }
    for (int i = gid; i < HID * HID; i += stride) {
        int n = i >> 7, k = i & (HID - 1);
        Wn[i] = (_Float16)W_hh[(2 * HID + n) * HID + k];
    }
    for (int i = gid; i < 4 * HID; i += stride) {
        int g = i >> 7, j = i & (HID - 1);
        float v;
        if (g == 0)      v = b_ih[j] + b_hh[j];
        else if (g == 1) v = b_ih[HID + j] + b_hh[HID + j];
        else if (g == 2) v = b_ih[2 * HID + j];
        else             v = b_hh[2 * HID + j];
        bb[i] = v;
    }
    for (int i = gid; i < PRED * EMB; i += stride) {
        int s = i / EMB, j = i - s * EMB;
        float v = b_emb[j];
        for (int f = 0; f < NFEAT; ++f) v += y_mark[s * NFEAT + f] * W_emb[f * EMB + j];
        ym[i] = v;
    }
}

// ------------------------------------------------------------- encoder ----
__global__ __launch_bounds__(256) void deepar_encoder(
    const float* __restrict__ x,     const float* __restrict__ x_mark,
    const float* __restrict__ W_emb, const float* __restrict__ b_emb,
    const float* __restrict__ W_ih,  const float* __restrict__ W_hh,
    const float* __restrict__ b_ih,  const float* __restrict__ b_hh,
    char* __restrict__ ws)
{
    __shared__ float red[256];
    __shared__ float xt[IN_DIM];
    __shared__ float hsh[HID];
    __shared__ float g[3 * HID];
    __shared__ float ghn[HID];
    __shared__ float s_inv;
    const int tid = threadIdx.x;

    float a = 0.f;
    for (int i = tid; i < CNTX; i += 256) a += fabsf(x[HIST - CNTX + i]);
    red[tid] = a;
    __syncthreads();
    for (int st = 128; st > 0; st >>= 1) {
        if (tid < st) red[tid] += red[tid + st];
        __syncthreads();
    }
    if (tid == 0) {
        float m = fmaxf(red[0] / (float)CNTX, 1e-5f);
        *(float*)(ws + WS_SC)  = m;
        *(float*)(ws + WS_INV) = 1.f / m;
        s_inv = 1.f / m;
    }
    if (tid < HID) hsh[tid] = 0.f;
    __syncthreads();
    const float inv = s_inv;

    for (int t = 0; t < CNTX; ++t) {
        for (int l = tid; l < NLAGS; l += 256)
            xt[l] = x[HIST - CNTX + t - 1 - l] * inv;
        for (int j = tid; j < EMB; j += 256) {
            float s = b_emb[j];
            for (int f = 0; f < NFEAT; ++f)
                s += x_mark[(HIST - CNTX + t) * NFEAT + f] * W_emb[f * EMB + j];
            xt[NLAGS + j] = s;
        }
        __syncthreads();
        for (int n = tid; n < 3 * HID; n += 256) {
            float gi = b_ih[n];
            const float* wr = &W_ih[n * IN_DIM];
            for (int k = 0; k < IN_DIM; ++k) gi += wr[k] * xt[k];
            float gh = b_hh[n];
            const float* wh = &W_hh[n * HID];
            for (int k = 0; k < HID; ++k) gh += wh[k] * hsh[k];
            if (n < 2 * HID) g[n] = gi + gh;
            else { g[n] = gi; ghn[n - 2 * HID] = gh; }
        }
        __syncthreads();
        if (tid < HID) {
            const int j = tid;
            float r  = 1.f / (1.f + __expf(-g[j]));
            float z  = 1.f / (1.f + __expf(-g[HID + j]));
            float nn = tanhf(g[2 * HID + j] + r * ghn[j]);
            hsh[j] = (1.f - z) * nn + z * hsh[j];
        }
        __syncthreads();
    }
    if (tid < HID) ((float*)(ws + WS_HLAST))[tid] = hsh[tid];
}

// ------------------------------------------------------------- decoder ----
__global__ __launch_bounds__(256) void deepar_decoder(
    const float* __restrict__ x, const float* __restrict__ W_proj,
    const float* __restrict__ b_proj, const char* __restrict__ ws,
    float* __restrict__ out)
{
    __shared__ __align__(16) _Float16 As[MBLK * XSTR];   // 23040 B, A-matrix (f16)
    __shared__ float buf[MBLK * BUFS];                   // 24576 B, lag ring (append-only)

    const int tid  = threadIdx.x;
    const int wv   = tid >> 5;        // wave 0..7: owns output cols j in [16w,16w+16)
    const int lane = tid & 31;
    const int t16  = lane & 15;
    const int hi   = lane >> 4;
    const int j    = wv * 16 + t16;   // this lane's h/gate column

    const float  sc     = *(const float*)(ws + WS_SC);
    const float  inv_sc = *(const float*)(ws + WS_INV);
    const float* h_last = (const float*)(ws + WS_HLAST);
    const float* ymf    = (const float*)(ws + WS_YM);
    const float* bb     = (const float*)(ws + WS_BB);
    const _Float16* Wn  = (const _Float16*)(ws + WS_WN);
    const _Float16* Wc  = (const _Float16*)(ws + WS_WC);

    const float bbr = bb[j], bbz = bb[HID + j], binN = bb[2 * HID + j], bhhN = bb[3 * HID + j];

    // per-wave base pointers for fragment loads (hoisted out of the K loop)
    const _Float16* Arow0 = &As[t16 * XSTR];
    const _Float16* Arow1 = &As[(16 + t16) * XSTR];
    const _Float16* WcR   = &Wc[(j) * KTOT];
    const _Float16* WcZ   = &Wc[(HID + j) * KTOT];
    const _Float16* WcN   = &Wc[(2 * HID + j) * KTOT];
    const _Float16* Wnj   = &Wn[(j) * HID];

    // init lag ring (identical rows: all paths start from the observed series)
    {
        const int m = tid >> 3;
        for (int k = (tid & 7); k < NLAGS; k += 8)
            buf[m * BUFS + k] = x[HIST - NLAGS + k];
    }
    // init h region of A (broadcast h_last) and the zero pad K∈[218,224)
    for (int idx = tid; idx < MBLK * HID; idx += 256) {
        int m = idx >> 7, k = idx & (HID - 1);
        As[m * XSTR + KH + k] = (_Float16)h_last[k];
    }
    for (int idx = tid; idx < MBLK * (KH - IN_DIM); idx += 256) {
        int m = idx / (KH - IN_DIM);
        int k = idx - m * (KH - IN_DIM);
        As[m * XSTR + IN_DIM + k] = (_Float16)0.f;
    }

    // h carried in WMMA-accumulator layout: hc[mt][e] <-> (m = mt*16+e+8*hi, n = j)
    v8f hc[2];
    {
        const float hl = h_last[j];
#pragma unroll
        for (int e = 0; e < 8; ++e) { hc[0][e] = hl; hc[1][e] = hl; }
    }

    for (int s = 0; s < PRED; ++s) {
        // (A) build lag + ym sections of A
        {
            const int m = tid >> 3;
            for (int k = (tid & 7); k < NLAGS; k += 8)
                As[m * XSTR + k] = (_Float16)(buf[m * BUFS + (s + NLAGS - 1 - k)] * inv_sc);
            for (int jj = (tid & 7); jj < EMB; jj += 8)
                As[m * XSTR + NLAGS + jj] = (_Float16)ymf[s * EMB + jj];
        }
        __syncthreads();

        // (B) fused GEMM over K=[0,352): r,z gates (x+h fused), i_n, and — for
        // kt>=7 (the h region) — the separate h@W_hh_n reusing the same A frags.
        v8f aR[2] = {{}, {}}, aZ[2] = {{}, {}}, aN[2] = {{}, {}}, g2[2] = {{}, {}};
#pragma unroll 2
        for (int kt = 0; kt < KTOT / 32; ++kt) {
            const int kb = kt * 32 + hi * 8;
            v16h a0 = load_frag(Arow0 + kb);
            v16h a1 = load_frag(Arow1 + kb);
            v16h bR = load_frag(WcR + kb);
            aR[0] = wmma16(a0, bR, aR[0]);  aR[1] = wmma16(a1, bR, aR[1]);
            v16h bZ = load_frag(WcZ + kb);
            aZ[0] = wmma16(a0, bZ, aZ[0]);  aZ[1] = wmma16(a1, bZ, aZ[1]);
            v16h bN = load_frag(WcN + kb);
            aN[0] = wmma16(a0, bN, aN[0]);  aN[1] = wmma16(a1, bN, aN[1]);
            if (kt >= (KH / 32)) {          // K in h region: also do n-gate h GEMM
                v16h bh = load_frag(Wnj + (kt - KH / 32) * 32 + hi * 8);
                g2[0] = wmma16(a0, bh, g2[0]);  g2[1] = wmma16(a1, bh, g2[1]);
            }
        }

        // (C) gates entirely in accumulator registers (same (m,n) across tiles)
#pragma unroll
        for (int mt = 0; mt < 2; ++mt) {
#pragma unroll
            for (int e = 0; e < 8; ++e) {
                float r  = 1.f / (1.f + __expf(-(aR[mt][e] + bbr)));
                float z  = 1.f / (1.f + __expf(-(aZ[mt][e] + bbz)));
                float nn = tanhf(aN[mt][e] + binN + r * (g2[mt][e] + bhhN));
                hc[mt][e] = (1.f - z) * nn + z * hc[mt][e];
            }
        }
        __syncthreads();   // everyone done reading old h region

        // (D) publish new h (f16) into A's h region for next step + projection
#pragma unroll
        for (int mt = 0; mt < 2; ++mt)
#pragma unroll
            for (int e = 0; e < 8; ++e)
                As[(mt * 16 + e + 8 * hi) * XSTR + KH + j] = (_Float16)hc[mt][e];
        __syncthreads();

        // (E) projection 128->3, Student-t draw (Bailey), feed back into lag ring
        {
            const int m = tid >> 3, p8 = tid & 7;
            float s0 = 0.f, s1 = 0.f, s2 = 0.f;
            for (int k = p8 * 16; k < p8 * 16 + 16; ++k) {
                float hv = (float)As[m * XSTR + KH + k];
                s0 += hv * W_proj[k * 3 + 0];
                s1 += hv * W_proj[k * 3 + 1];
                s2 += hv * W_proj[k * 3 + 2];
            }
#pragma unroll
            for (int off = 4; off > 0; off >>= 1) {
                s0 += __shfl_xor(s0, off);
                s1 += __shfl_xor(s1, off);
                s2 += __shfl_xor(s2, off);
            }
            if (p8 == 0) {
                float df  = 2.f + softplusf(s0 + b_proj[0]);
                float loc = s1 + b_proj[1];
                float sig = softplusf(s2 + b_proj[2]);
                const uint32_t row = blockIdx.x * MBLK + m;
                uint64_t r0 = splitmix64((uint64_t)row * (uint64_t)PRED + (uint64_t)s + 0x42ull);
                float u1 = (float)(uint32_t)(r0 >> 40) * (1.f / 16777216.f) + (0.5f / 16777216.f);
                float u2 = (float)((uint32_t)(r0 >> 16) & 0xFFFFFFu) * (1.f / 16777216.f) + (0.5f / 16777216.f);
                float tt = sqrtf(df * (__powf(u1, -2.f / df) - 1.f)) * __cosf(6.28318530718f * u2);
                float samp = (loc + sig * tt) * sc;
                buf[m * BUFS + NLAGS + s] = samp;
                out[(size_t)row * PRED + s] = samp;
            }
        }
        __syncthreads();
    }
}

// ------------------------------------------------------------- launch -----
extern "C" void kernel_launch(void* const* d_in, const int* in_sizes, int n_in,
                              void* d_out, int out_size, void* d_ws, size_t ws_size,
                              hipStream_t stream) {
    const float* x      = (const float*)d_in[0];
    const float* x_mark = (const float*)d_in[1];
    const float* y_mark = (const float*)d_in[2];
    const float* W_emb  = (const float*)d_in[3];
    const float* b_emb  = (const float*)d_in[4];
    const float* W_ih   = (const float*)d_in[5];
    const float* W_hh   = (const float*)d_in[6];
    const float* b_ih   = (const float*)d_in[7];
    const float* b_hh   = (const float*)d_in[8];
    const float* W_proj = (const float*)d_in[9];
    const float* b_proj = (const float*)d_in[10];
    char*  ws  = (char*)d_ws;
    float* out = (float*)d_out;

    deepar_prep<<<64, 256, 0, stream>>>(y_mark, W_emb, b_emb, W_ih, W_hh, b_ih, b_hh, ws);
    deepar_encoder<<<1, 256, 0, stream>>>(x, x_mark, W_emb, b_emb, W_ih, W_hh, b_ih, b_hh, ws);
    deepar_decoder<<<NSAMP / MBLK, 256, 0, stream>>>(x, W_proj, b_proj, ws, out);
}
